// PVPSegNet_11776800325758
// MI455X (gfx1250) — compile-verified, matrix operand order checked by reading, not run on hardware
//
#include <hip/hip_runtime.h>
#include <hip/hip_bf16.h>

typedef __attribute__((ext_vector_type(16))) _Float16 v16h;
typedef __attribute__((ext_vector_type(8)))  _Float16 v8h;
typedef __attribute__((ext_vector_type(8)))  float    v8f;

#define EPSF 1e-5f
static constexpr int Bn = 4, NPTS = 2048, NCLS = 50, NCAT = 16, WGRID = 32;
static constexpr int Gv = WGRID * WGRID * WGRID;   // 32768

// ---------------------------------------------------------------------------
// 1. Point MLP (6->64 bn relu, 64->64) + analytic nearest-grid index
// ---------------------------------------------------------------------------
__global__ void k_point_mlp(const float* __restrict__ x,      // (B,N,6)
                            const float* __restrict__ w1, const float* __restrict__ b1,
                            const float* __restrict__ g1, const float* __restrict__ be1,
                            const float* __restrict__ m1, const float* __restrict__ v1,
                            const float* __restrict__ w2, const float* __restrict__ b2,
                            float* __restrict__ pf, int* __restrict__ gidx)
{
    __shared__ float sW1[64 * 6], sB1[64], sG[64], sBe[64], sM[64], sV[64];
    __shared__ float sW2[64 * 64], sB2[64];
    int t = threadIdx.x;
    for (int i = t; i < 64 * 6; i += blockDim.x) sW1[i] = w1[i];
    for (int i = t; i < 64; i += blockDim.x) {
        sB1[i] = b1[i]; sG[i] = g1[i]; sBe[i] = be1[i];
        sM[i] = m1[i];  sV[i] = v1[i]; sB2[i] = b2[i];
    }
    for (int i = t; i < 64 * 64; i += blockDim.x) sW2[i] = w2[i];
    __syncthreads();

    int p = blockIdx.x * blockDim.x + t;
    if (p >= Bn * NPTS) return;

    float xi[6];
    for (int c = 0; c < 6; ++c) xi[c] = x[p * 6 + c];

    // nearest grid cell on regular 32^3 grid (_grid_points layout: g=u*1024+v*32+k)
    int u = (int)floorf((xi[0] + 1.f) * 16.f); u = u < 0 ? 0 : (u > 31 ? 31 : u);
    int k = (int)floorf((xi[1] + 1.f) * 16.f); k = k < 0 ? 0 : (k > 31 ? 31 : k);
    int v = (int)floorf((1.f - xi[2]) * 16.f); v = v < 0 ? 0 : (v > 31 ? 31 : v);
    gidx[p] = u * 1024 + v * 32 + k;

    float h[64];
    for (int o = 0; o < 64; ++o) {
        float a = sB1[o];
        for (int c = 0; c < 6; ++c) a += xi[c] * sW1[o * 6 + c];
        a = (a - sM[o]) * (sG[o] * rsqrtf(sV[o] + EPSF)) + sBe[o];
        h[o] = a > 0.f ? a : 0.f;
    }
    for (int o = 0; o < 64; ++o) {
        float a = sB2[o];
        for (int c = 0; c < 64; ++c) a += h[c] * sW2[o * 64 + c];
        pf[p * 64 + o] = a;
    }
}

// ---------------------------------------------------------------------------
// 2. Voxelization: scatter mean
// ---------------------------------------------------------------------------
__global__ void k_scatter(const float* __restrict__ pf, const int* __restrict__ gidx,
                          float* __restrict__ sums, float* __restrict__ cnts)
{
    int i = blockIdx.x * blockDim.x + threadIdx.x;   // over B*N*64
    if (i >= Bn * NPTS * 64) return;
    int c = i & 63, p = i >> 6;
    int b = p / NPTS;
    int fg = b * Gv + gidx[p];
    atomicAdd(&sums[(long)fg * 64 + c], pf[i]);
    if (c == 0) atomicAdd(&cnts[fg], 1.f);
}

__global__ void k_voxnorm(const float* __restrict__ sums, const float* __restrict__ cnts,
                          float* __restrict__ p2v, _Float16* __restrict__ p2vh)
{
    long i = blockIdx.x * (long)blockDim.x + threadIdx.x;   // layout (b,g,c)
    if (i >= (long)Bn * Gv * 64) return;
    int c = (int)(i & 63);
    long bg = i >> 6;
    int g = (int)(bg % Gv), b = (int)(bg / Gv);
    float val = sums[i] / fmaxf(cnts[bg], 1.f);
    long o = ((long)(b * 64 + c)) * Gv + g;   // (B,64,32,32,32)
    p2v[o] = val; p2vh[o] = (_Float16)val;
}

// ---------------------------------------------------------------------------
// 3. Weight prep: f32 -> f16, pre-swizzled into the per-lane B-fragment layout
//    of V_WMMA_F32_16X16X32_F16 (lane<16: K{0..7,16..23}, lane>=16: K{8..15,24..31})
//    Layout: Wsw[(((ks*NT + nt)*32 + lane)*16 + e]
// ---------------------------------------------------------------------------
__global__ void k_prep_w(const float* __restrict__ w, _Float16* __restrict__ ws,
                         int Cin, int Cout, int deconv)
{
    long i = blockIdx.x * (long)blockDim.x + threadIdx.x;
    long total = (long)Cin * 64 * Cout;
    if (i >= total) return;
    int  e    = (int)(i & 15);
    int  lane = (int)((i >> 4) & 31);
    long r    = i >> 9;
    int  NT   = Cout >> 4;
    int  nt   = (int)(r % NT);
    long ks   = r / NT;
    int  kof  = (e < 8 ? e : e + 8) + ((lane >> 4) << 3);
    int  kk   = (int)(ks * 32 + kof);
    int  ci = kk >> 6, tap = kk & 63;
    int  td = tap >> 4, th = (tap >> 2) & 3, tw = tap & 3;
    int  co = nt * 16 + (lane & 15);
    float val;
    if (deconv)   // reference flips spatially + transposes (Cin,Cout,4,4,4)
        val = w[((long)ci * Cout + co) * 64 + (3 - td) * 16 + (3 - th) * 4 + (3 - tw)];
    else          // (Cout,Cin,4,4,4)
        val = w[((long)co * Cin + ci) * 64 + td * 16 + th * 4 + tw];
    ws[i] = (_Float16)val;
}

// ---------------------------------------------------------------------------
// 4. Implicit-GEMM conv3d / deconv3d via WMMA f16 -> f32
//    MODE 0: conv k=4 pad=2 stride=1 (DOUT = DIN+1), epilogue bias+BN+LeakyReLU(0.2)
//    MODE 1: transposed conv k=4 stride=2 (DOUT = 2*DIN), epilogue BN+ReLU+skip
//    Block: 128 threads = 4 waves; 32-voxel M tile, each wave a 16-wide N tile
//    doing 2 WMMAs per B fragment. Gather addressing fully hoisted; LDS A-tile
//    double buffered (row stride 40 f16 -> conflict-free b128 reads).
// ---------------------------------------------------------------------------
template <int CIN, int DIN, int DOUT, int MODE>
__global__ __launch_bounds__(128)
void k_conv_wmma(const _Float16* __restrict__ X,    // (B,CIN,DIN^3) f16
                 const _Float16* __restrict__ Wsw,  // swizzled weights
                 const float* __restrict__ bias,
                 const float* __restrict__ bg, const float* __restrict__ bb,
                 const float* __restrict__ bm, const float* __restrict__ bv,
                 const float* __restrict__ skip,    // (B,Cout,DOUT^3) or null
                 float* __restrict__ Yf, _Float16* __restrict__ Yh, int Cout)
{
    constexpr int M3 = DOUT * DOUT * DOUT;
    constexpr int KS = CIN * 2;                 // K = CIN*64, steps of 32
    constexpr int LD = 40;                      // LDS row stride (f16), 80B
    __shared__ __align__(32) _Float16 sA[2][32][LD];

    const int tid = threadIdx.x, lane = tid & 31, wave = tid >> 5;
    const int mtile = blockIdx.x, b = blockIdx.z;
    const int NT  = Cout >> 4;
    const int ntg = blockIdx.y * 4 + wave;
    const _Float16* Xb = X + (long)b * CIN * DIN * DIN * DIN;

    // ---- hoisted gather state: column kk fixed per thread, 8 rows = wave+4j,
    //      tap decode alternates only with ks parity; ci = ks>>1 ----
    const int kk = tid & 31;
    int      soff[2][8];
    unsigned okbits = 0;
#pragma unroll
    for (int p2 = 0; p2 < 2; ++p2) {
        int tap = kk + 32 * p2;
        int td = tap >> 4, th = (tap >> 2) & 3, tw = tap & 3;
#pragma unroll
        for (int j = 0; j < 8; ++j) {
            int m  = mtile * 32 + wave + j * 4;
            int od = m / (DOUT * DOUT), rm = m % (DOUT * DOUT);
            int oh = rm / DOUT, ow = rm % DOUT;
            bool ok = (m < M3);
            int id = 0, ih = 0, iw = 0;
            if (MODE == 0) {
                id = od + td - 2; ih = oh + th - 2; iw = ow + tw - 2;
                ok = ok && (unsigned)id < (unsigned)DIN && (unsigned)ih < (unsigned)DIN &&
                     (unsigned)iw < (unsigned)DIN;
            } else {
                int pd = od + td - 2, ph = oh + th - 2, pw = ow + tw - 2;
                ok = ok && pd >= 0 && ph >= 0 && pw >= 0 &&
                     !(pd & 1) && !(ph & 1) && !(pw & 1);
                id = pd >> 1; ih = ph >> 1; iw = pw >> 1;
                ok = ok && id < DIN && ih < DIN && iw < DIN;
            }
            soff[p2][j] = ok ? (id * DIN + ih) * DIN + iw : 0;
            if (ok) okbits |= 1u << (p2 * 8 + j);
        }
    }

    auto fill = [&](int ks, int buf) {
        const int p2 = ks & 1;
        const _Float16* Xc = Xb + (long)(ks >> 1) * (DIN * DIN * DIN);
#pragma unroll
        for (int j = 0; j < 8; ++j) {
            _Float16 ld = Xc[soff[p2][j]];                  // offset 0 when invalid
            _Float16 v  = ((okbits >> (p2 * 8 + j)) & 1u) ? ld : (_Float16)0.f;
            sA[buf][wave + j * 4][kk] = v;
        }
    };

    fill(0, 0);
    __syncthreads();

    v8f acc0 = {}, acc1 = {};
    const int row = lane & 15, k0 = (lane >> 4) << 3;
    for (int ks = 0; ks < KS; ++ks) {
        const int buf = ks & 1;
        if (ks + 1 < KS) fill(ks + 1, buf ^ 1);

        // A fragments (two 16-row halves of the 32-row tile)
        union { v16h v; v8h p[2]; } a0, a1;
        a0.p[0] = *(const v8h*)&sA[buf][row][k0];
        a0.p[1] = *(const v8h*)&sA[buf][row][16 + k0];
        a1.p[0] = *(const v8h*)&sA[buf][16 + row][k0];
        a1.p[1] = *(const v8h*)&sA[buf][16 + row][16 + k0];

        // B fragment: pre-swizzled, one contiguous 32B load per lane
        const _Float16* wpt = Wsw + (((long)ks * NT + ntg) * 32 + lane) * 16;
        v16h bf = *(const v16h*)wpt;
        if (ks + 1 < KS)
            __builtin_prefetch(Wsw + (((long)(ks + 1) * NT + ntg) * 32 + lane) * 16, 0, 1);

        acc0 = __builtin_amdgcn_wmma_f32_16x16x32_f16(
            false, a0.v, false, bf, (short)0, acc0, false, false);
        acc1 = __builtin_amdgcn_wmma_f32_16x16x32_f16(
            false, a1.v, false, bf, (short)0, acc1, false, false);
        __syncthreads();
    }

    // epilogue: lane holds column n; acc0 rows 0..15, acc1 rows 16..31 of tile
    int n  = lane & 15;
    int co = blockIdx.y * 64 + wave * 16 + n;
    float sc = bg[co] * rsqrtf(bv[co] + EPSF);
    float sh = bb[co] - bm[co] * sc;
    float bi = bias ? bias[co] : 0.f;
    int mbase = mtile * 32 + ((lane >> 4) << 3);
#pragma unroll
    for (int half = 0; half < 2; ++half) {
        const v8f& acc = half ? acc1 : acc0;
#pragma unroll
        for (int r = 0; r < 8; ++r) {
            int m = mbase + half * 16 + r;
            if (m < M3) {
                float val = (acc[r] + bi) * sc + sh;
                long oi = ((long)(b * Cout + co)) * M3 + m;
                if (MODE == 0) {
                    val = val > 0.f ? val : 0.2f * val;          // LeakyReLU(0.2)
                } else {
                    val = val > 0.f ? val : 0.f;                 // ReLU
                    val += skip[oi];                             // residual
                }
                Yf[oi] = val;
                if (Yh) Yh[oi] = (_Float16)val;
            }
        }
    }
}

// ---------------------------------------------------------------------------
// 5. 2x2x2 max-pool (VALID)
// ---------------------------------------------------------------------------
__global__ void k_maxpool(const float* __restrict__ X, float* __restrict__ Yf,
                          _Float16* __restrict__ Yh, int C, int Din, int Dout)
{
    long i = blockIdx.x * (long)blockDim.x + threadIdx.x;
    long tot = (long)Bn * C * Dout * Dout * Dout;
    if (i >= tot) return;
    int ow = (int)(i % Dout); long r = i / Dout;
    int oh = (int)(r % Dout); r /= Dout;
    int od = (int)(r % Dout); r /= Dout;           // r = b*C + c
    const float* Xp = X + r * (long)Din * Din * Din;
    float mx = -1e30f;
    for (int dd = 0; dd < 2; ++dd)
        for (int hh = 0; hh < 2; ++hh)
            for (int ww = 0; ww < 2; ++ww)
                mx = fmaxf(mx, Xp[((long)(2 * od + dd) * Din + (2 * oh + hh)) * Din + (2 * ow + ww)]);
    Yf[i] = mx; Yh[i] = (_Float16)mx;
}

// ---------------------------------------------------------------------------
// 6. FC (M=4 -> VALU): y = relu(A @ W^T + b) [+ skip]
// ---------------------------------------------------------------------------
__global__ void k_fc(const float* __restrict__ A, const float* __restrict__ W,
                     const float* __restrict__ bias, const float* __restrict__ skip,
                     float* __restrict__ Yf, _Float16* __restrict__ Yh, int K, int N)
{
    int i = blockIdx.x * blockDim.x + threadIdx.x;
    if (i >= Bn * N) return;
    int n = i % N, b = i / N;
    const float* a = A + (long)b * K;
    const float* w = W + (long)n * K;
    float s = bias[n];
    for (int k = 0; k < K; ++k) s += a[k] * w[k];
    s = s > 0.f ? s : 0.f;
    if (skip) s += skip[i];
    Yf[i] = s;
    if (Yh) Yh[i] = (_Float16)s;
}

// ---------------------------------------------------------------------------
// 7. Point head: gather voxel feats, concat [pf|gather|cls], fc1+bn+relu
// ---------------------------------------------------------------------------
__global__ void k_head1(const float* __restrict__ pf, const int* __restrict__ gidx,
                        const float* __restrict__ v32r, const float* __restrict__ cls,
                        const float* __restrict__ w1,
                        const float* __restrict__ g, const float* __restrict__ be,
                        const float* __restrict__ m, const float* __restrict__ v,
                        float* __restrict__ hout)
{
    __shared__ float cat[144];
    int p = blockIdx.x, t = threadIdx.x;      // 64 threads, one point per block
    int b = p / NPTS;
    int gi = gidx[p];
    cat[t] = pf[(long)p * 64 + t];
    cat[64 + t] = v32r[((long)(b * 64 + t)) * Gv + gi];
    if (t < 16) cat[128 + t] = cls[b * 16 + t];
    __syncthreads();
    float a = 0.f;
    const float* wr = w1 + t * 144;
    for (int c = 0; c < 144; ++c) a += cat[c] * wr[c];
    a = (a - m[t]) * (g[t] * rsqrtf(v[t] + EPSF)) + be[t];
    hout[(long)p * 64 + t] = a > 0.f ? a : 0.f;
}

__global__ void k_head2(const float* __restrict__ h, const float* __restrict__ w2,
                        const float* __restrict__ b2, float* __restrict__ out)
{
    int i = blockIdx.x * blockDim.x + threadIdx.x;
    if (i >= Bn * NPTS * NCLS) return;
    int cl = i % NCLS, p = i / NCLS;
    float s = b2[cl];
    const float* hr = h + (long)p * 64;
    const float* wr = w2 + cl * 64;
    for (int c = 0; c < 64; ++c) s += hr[c] * wr[c];
    out[i] = s;
}

// ---------------------------------------------------------------------------
// launcher
// ---------------------------------------------------------------------------
extern "C" void kernel_launch(void* const* d_in, const int* in_sizes, int n_in,
                              void* d_out, int out_size, void* d_ws, size_t ws_size,
                              hipStream_t stream)
{
    // input order: x, cls_, then params dict (bn tuples flatten as g,b,m,v)
    const float* x       = (const float*)d_in[0];
    const float* cls     = (const float*)d_in[1];
    const float* p2v_w1  = (const float*)d_in[2];
    const float* p2v_b1  = (const float*)d_in[3];
    const float* p2v_bn[4] = {(const float*)d_in[4], (const float*)d_in[5],
                              (const float*)d_in[6], (const float*)d_in[7]};
    const float* p2v_w2  = (const float*)d_in[8];
    const float* p2v_b2  = (const float*)d_in[9];
    const float* cw[4];  const float* cb[4];  const float* cbn[4][4];
    for (int l = 0; l < 4; ++l) {
        int base = 10 + l * 6;
        cw[l] = (const float*)d_in[base];
        cb[l] = (const float*)d_in[base + 1];
        for (int j = 0; j < 4; ++j) cbn[l][j] = (const float*)d_in[base + 2 + j];
    }
    const float* fc5_w = (const float*)d_in[34];
    const float* fc5_b = (const float*)d_in[35];
    const float* fc6_w = (const float*)d_in[36];
    const float* fc6_b = (const float*)d_in[37];
    const float* dw[4];  const float* dbn[4][4];
    for (int l = 0; l < 4; ++l) {
        int base = 38 + l * 5;
        dw[l] = (const float*)d_in[base];
        for (int j = 0; j < 4; ++j) dbn[l][j] = (const float*)d_in[base + 1 + j];
    }
    const float* fc_w1 = (const float*)d_in[58];
    const float* fc_bn[4] = {(const float*)d_in[59], (const float*)d_in[60],
                             (const float*)d_in[61], (const float*)d_in[62]};
    const float* fc_w2 = (const float*)d_in[63];
    const float* fc_b2 = (const float*)d_in[64];

    // -------- workspace bump allocator --------
    char* wp = (char*)d_ws;
    auto alloc = [&](size_t bytes) -> void* {
        void* r = (void*)wp;
        wp += (bytes + 255) & ~((size_t)255);
        return r;
    };
    const long NP = (long)Bn * NPTS;
    float* pf    = (float*)alloc(NP * 64 * 4);
    int*   gidx  = (int*)  alloc(NP * 4);
    float* sums  = (float*)alloc((long)Bn * Gv * 64 * 4);
    float* cnts  = (float*)alloc((long)Bn * Gv * 4);
    float* p2vf  = (float*)alloc((long)Bn * 64 * Gv * 4);
    _Float16* p2vh = (_Float16*)alloc((long)Bn * 64 * Gv * 2);
    float* c1o   = (float*)alloc((long)Bn * 64  * 35937 * 4);
    float* v16f  = (float*)alloc((long)Bn * 64  * 4096 * 4);
    _Float16* v16h_ = (_Float16*)alloc((long)Bn * 64 * 4096 * 2);
    float* c2o   = (float*)alloc((long)Bn * 128 * 4913 * 4);
    float* v8f_  = (float*)alloc((long)Bn * 128 * 512 * 4);
    _Float16* v8h_ = (_Float16*)alloc((long)Bn * 128 * 512 * 2);
    float* c3o   = (float*)alloc((long)Bn * 256 * 729 * 4);
    float* v4f_  = (float*)alloc((long)Bn * 256 * 64 * 4);
    _Float16* v4h_ = (_Float16*)alloc((long)Bn * 256 * 64 * 2);
    float* c4o   = (float*)alloc((long)Bn * 512 * 125 * 4);
    float* v2f_  = (float*)alloc((long)Bn * 512 * 8 * 4);
    _Float16* v2h_ = (_Float16*)alloc((long)Bn * 512 * 8 * 2);
    float* feat  = (float*)alloc((long)Bn * 2048 * 4);
    float* v2rf  = (float*)alloc((long)Bn * 4096 * 4);
    _Float16* v2rh = (_Float16*)alloc((long)Bn * 4096 * 2);
    float* v4rf  = (float*)alloc((long)Bn * 256 * 64 * 4);
    _Float16* v4rh = (_Float16*)alloc((long)Bn * 256 * 64 * 2);
    float* v8rf  = (float*)alloc((long)Bn * 128 * 512 * 4);
    _Float16* v8rh = (_Float16*)alloc((long)Bn * 128 * 512 * 2);
    float* v16rf = (float*)alloc((long)Bn * 64 * 4096 * 4);
    _Float16* v16rh = (_Float16*)alloc((long)Bn * 64 * 4096 * 2);
    float* v32rf = (float*)alloc((long)Bn * 64 * Gv * 4);
    float* hbuf  = (float*)alloc(NP * 64 * 4);
    _Float16* Wc[4]; _Float16* Wd[4];
    const int cCin[4] = {64, 64, 128, 256}, cCout[4] = {64, 128, 256, 512};
    const int dCin[4] = {512, 256, 128, 64}, dCout[4] = {256, 128, 64, 64};
    for (int l = 0; l < 4; ++l) Wc[l] = (_Float16*)alloc((long)cCin[l] * 64 * cCout[l] * 2);
    for (int l = 0; l < 4; ++l) Wd[l] = (_Float16*)alloc((long)dCin[l] * 64 * dCout[l] * 2);

    // -------- pipeline --------
    hipMemsetAsync(sums, 0, (long)Bn * Gv * 64 * 4, stream);
    hipMemsetAsync(cnts, 0, (long)Bn * Gv * 4, stream);

    k_point_mlp<<<(NP + 255) / 256, 256, 0, stream>>>(
        x, p2v_w1, p2v_b1, p2v_bn[0], p2v_bn[1], p2v_bn[2], p2v_bn[3],
        p2v_w2, p2v_b2, pf, gidx);
    k_scatter<<<(int)((NP * 64 + 255) / 256), 256, 0, stream>>>(pf, gidx, sums, cnts);
    k_voxnorm<<<(int)(((long)Bn * Gv * 64 + 255) / 256), 256, 0, stream>>>(sums, cnts, p2vf, p2vh);

    for (int l = 0; l < 4; ++l) {
        long tot = (long)cCin[l] * 64 * cCout[l];
        k_prep_w<<<(int)((tot + 255) / 256), 256, 0, stream>>>(cw[l], Wc[l], cCin[l], cCout[l], 0);
        tot = (long)dCin[l] * 64 * dCout[l];
        k_prep_w<<<(int)((tot + 255) / 256), 256, 0, stream>>>(dw[l], Wd[l], dCin[l], dCout[l], 1);
    }

    // encoder: conv(+bn+lrelu) then maxpool  (M-tile = 32 voxels now)
    k_conv_wmma<64, 32, 33, 0><<<dim3(1124, 1, Bn), 128, 0, stream>>>(
        p2vh, Wc[0], cb[0], cbn[0][0], cbn[0][1], cbn[0][2], cbn[0][3],
        nullptr, c1o, nullptr, 64);
    k_maxpool<<<(int)(((long)Bn * 64 * 4096 + 255) / 256), 256, 0, stream>>>(c1o, v16f, v16h_, 64, 33, 16);

    k_conv_wmma<64, 16, 17, 0><<<dim3(154, 2, Bn), 128, 0, stream>>>(
        v16h_, Wc[1], cb[1], cbn[1][0], cbn[1][1], cbn[1][2], cbn[1][3],
        nullptr, c2o, nullptr, 128);
    k_maxpool<<<(int)(((long)Bn * 128 * 512 + 255) / 256), 256, 0, stream>>>(c2o, v8f_, v8h_, 128, 17, 8);

    k_conv_wmma<128, 8, 9, 0><<<dim3(23, 4, Bn), 128, 0, stream>>>(
        v8h_, Wc[2], cb[2], cbn[2][0], cbn[2][1], cbn[2][2], cbn[2][3],
        nullptr, c3o, nullptr, 256);
    k_maxpool<<<(int)(((long)Bn * 256 * 64 + 255) / 256), 256, 0, stream>>>(c3o, v4f_, v4h_, 256, 9, 4);

    k_conv_wmma<256, 4, 5, 0><<<dim3(4, 8, Bn), 128, 0, stream>>>(
        v4h_, Wc[3], cb[3], cbn[3][0], cbn[3][1], cbn[3][2], cbn[3][3],
        nullptr, c4o, nullptr, 512);
    k_maxpool<<<(int)(((long)Bn * 512 * 8 + 255) / 256), 256, 0, stream>>>(c4o, v2f_, v2h_, 512, 5, 2);

    // bottleneck FCs
    k_fc<<<(Bn * 2048 + 255) / 256, 256, 0, stream>>>(v2f_, fc5_w, fc5_b, nullptr, feat, nullptr, 4096, 2048);
    k_fc<<<(Bn * 4096 + 255) / 256, 256, 0, stream>>>(feat, fc6_w, fc6_b, v2f_, v2rf, v2rh, 2048, 4096);

    // decoder: deconv(+bn+relu+skip)
    k_conv_wmma<512, 2, 4, 1><<<dim3(2, 4, Bn), 128, 0, stream>>>(
        v2rh, Wd[0], nullptr, dbn[0][0], dbn[0][1], dbn[0][2], dbn[0][3],
        v4f_, v4rf, v4rh, 256);
    k_conv_wmma<256, 4, 8, 1><<<dim3(16, 2, Bn), 128, 0, stream>>>(
        v4rh, Wd[1], nullptr, dbn[1][0], dbn[1][1], dbn[1][2], dbn[1][3],
        v8f_, v8rf, v8rh, 128);
    k_conv_wmma<128, 8, 16, 1><<<dim3(128, 1, Bn), 128, 0, stream>>>(
        v8rh, Wd[2], nullptr, dbn[2][0], dbn[2][1], dbn[2][2], dbn[2][3],
        v16f, v16rf, v16rh, 64);
    k_conv_wmma<64, 16, 32, 1><<<dim3(1024, 1, Bn), 128, 0, stream>>>(
        v16rh, Wd[3], nullptr, dbn[3][0], dbn[3][1], dbn[3][2], dbn[3][3],
        p2vf, v32rf, nullptr, 64);

    // per-point head
    k_head1<<<(int)NP, 64, 0, stream>>>(pf, gidx, v32rf, cls, fc_w1,
                                        fc_bn[0], fc_bn[1], fc_bn[2], fc_bn[3], hbuf);
    k_head2<<<(int)((NP * NCLS + 255) / 256), 256, 0, stream>>>(hbuf, fc_w2, fc_b2, (float*)d_out);
}
